// MeshAutoencoder_37838661878234
// MI455X (gfx1250) — matrix operand order repaired and checked by reading, not compiled, not run on hardware
//
#include <hip/hip_runtime.h>

typedef __attribute__((ext_vector_type(16))) _Float16 v16h;
typedef __attribute__((ext_vector_type(8)))  _Float16 v8h;
typedef __attribute__((ext_vector_type(8)))  float    v8f;
typedef _Float16 f16;

constexpr int cB  = 2;
constexpr int cNV = 8192;
constexpr int cNF = 16384;
constexpr int cNE = 49152;
constexpr int cM  = cB * cNF;   // 32768 faces total
constexpr int cMV = cB * cNV;   // 16384 vertices total
constexpr float cPI = 3.14159265358979323846f;

#define GF_ACC  1
#define GF_RELU 2
#define GF_SILU 4
#define RN_L2     1
#define RN_SILULN 2

// ---------------------------------------------------------------------------
// utility kernels
// ---------------------------------------------------------------------------
__global__ void k_zero(float* __restrict__ p, long long n) {
  long long i = (long long)blockIdx.x * blockDim.x + threadIdx.x;
  long long s = (long long)gridDim.x * blockDim.x;
  for (; i < n; i += s) p[i] = 0.f;
}

// weight repack: src [K][N] f32 (row-major, x@W layout) -> dst [Npad][Kpad] f16 (W^T, zero padded)
__global__ void k_wt(const float* __restrict__ s, f16* __restrict__ d,
                     int K, int N, int Kpad, int Npad) {
  long long n = (long long)Kpad * Npad;
  long long i = (long long)blockIdx.x * blockDim.x + threadIdx.x;
  long long st = (long long)gridDim.x * blockDim.x;
  for (; i < n; i += st) {
    int nn = (int)(i / Kpad), k = (int)(i % Kpad);
    d[i] = (nn < N && k < K) ? (f16)s[(long long)k * N + nn] : (f16)0.f;
  }
}

// conv weight repack: src [O][I][T] (OIH) -> dst [T][O][I]  (per-tap W^T tiles)
__global__ void k_convwt(const float* __restrict__ s, f16* __restrict__ d, int O, int I, int T) {
  long long n = (long long)O * I * T;
  long long i = (long long)blockIdx.x * blockDim.x + threadIdx.x;
  long long st = (long long)gridDim.x * blockDim.x;
  for (; i < n; i += st) {
    int t = (int)(i / ((long long)O * I));
    int rem = (int)(i % ((long long)O * I));
    int o = rem / I, ii = rem % I;
    d[i] = (f16)s[((long long)o * I + ii) * T + t];
  }
}

// ---------------------------------------------------------------------------
// WMMA GEMM:  C[M,N] (f32) = act( (acc? C : bias) + A[M,K](f16) @ W[K,N] )
// Bt is the transposed weight [N][K] (f16, row-major) so each B fragment is
// two contiguous 128-bit loads. K must be a multiple of 32 (host pads).
// Each wave computes a 32x64 strip: 2 A fragments x 4 B fragments -> 8 WMMA.
// Optional row shift (conv tap) within per-batch sequences of length seqLen,
// optional f16 mirror of the output (feeds the next GEMM).
// ---------------------------------------------------------------------------
__global__ __launch_bounds__(128) void k_gemm(
    const f16* __restrict__ A, const f16* __restrict__ Bt,
    const float* __restrict__ bias, float* __restrict__ C, f16* __restrict__ C16,
    int M, int N, int K, int shift, int seqLen, int flags)
{
  const int lane = threadIdx.x & 31;
  const int wave = threadIdx.x >> 5;
  const int m0 = (blockIdx.y * 4 + wave) * 32;
  const int n0 = blockIdx.x * 64;
  if (m0 >= M) return;                   // wave-uniform: EXEC all-ones at WMMA
  const int hl  = lane >> 4;             // half-wave select
  const int l15 = lane & 15;

  // two A fragment rows (with conv-tap shift + per-batch boundary mask)
  bool rv[2]; const f16* Ar[2];
#pragma unroll
  for (int u = 0; u < 2; ++u) {
    int r = m0 + u * 16 + l15;
    long long srow;
    if (seqLen > 0) {
      int bb = r / seqLen, nn = r - bb * seqLen;
      int ns = nn + shift;
      rv[u] = (ns >= 0) && (ns < seqLen);
      srow = (long long)bb * seqLen + ns;
    } else { rv[u] = (r < M); srow = r; }
    Ar[u] = A + srow * (long long)K + hl * 8;
  }
  // four B column pointers
  bool cv[4]; const f16* Bp[4];
#pragma unroll
  for (int t = 0; t < 4; ++t) {
    int col = n0 + t * 16 + l15;
    cv[t] = (col < N);
    Bp[t] = Bt + (long long)(cv[t] ? col : 0) * K + hl * 8;
  }

  v8f acc[2][4] = {};
  for (int k0 = 0; k0 < K; k0 += 32) {
    __builtin_prefetch(Ar[0] + k0 + 64, 0, 0);   // global_prefetch_b8 on the A stream
    v16h a[2];
#pragma unroll
    for (int u = 0; u < 2; ++u) {
      if (rv[u]) {
        v8h lo = *(const v8h*)(Ar[u] + k0);
        v8h hi = *(const v8h*)(Ar[u] + k0 + 16);
        a[u] = __builtin_shufflevector(lo, hi, 0,1,2,3,4,5,6,7,8,9,10,11,12,13,14,15);
      } else {
        a[u] = (v16h){};
      }
    }
#pragma unroll
    for (int t = 0; t < 4; ++t) {
      v16h bf;
      if (cv[t]) {
        v8h b0 = *(const v8h*)(Bp[t] + k0);
        v8h b1 = *(const v8h*)(Bp[t] + k0 + 16);
        bf = __builtin_shufflevector(b0, b1, 0,1,2,3,4,5,6,7,8,9,10,11,12,13,14,15);
      } else {
        bf = (v16h){};
      }
      acc[0][t] = __builtin_amdgcn_wmma_f32_16x16x32_f16(
          false, a[0], false, bf, (short)0, acc[0][t], false, false);
      acc[1][t] = __builtin_amdgcn_wmma_f32_16x16x32_f16(
          false, a[1], false, bf, (short)0, acc[1][t], false, false);
    }
  }

#pragma unroll
  for (int u = 0; u < 2; ++u) {
    const int rr0 = m0 + u * 16 + hl * 8;
#pragma unroll
    for (int t = 0; t < 4; ++t) {
      const int col = n0 + t * 16 + l15;
      if (col >= N) continue;
#pragma unroll
      for (int e = 0; e < 8; ++e) {
        int rr = rr0 + e;
        if (rr >= M) continue;
        long long idx = (long long)rr * N + col;
        float v = acc[u][t][e];
        if (flags & GF_ACC)       v += C[idx];
        else if (bias)            v += bias[col];
        if (flags & GF_RELU)      v = v > 0.f ? v : 0.f;
        else if (flags & GF_SILU) v = v / (1.f + __expf(-v));
        C[idx] = v;
        if (C16) C16[idx] = (f16)v;
      }
    }
  }
}

// ---------------------------------------------------------------------------
// face geometric features -> f16 embedding rows [cM, 832]
// 8 faces per block, 32 lanes cooperate on the 832 embedding writes
// ---------------------------------------------------------------------------
__device__ __forceinline__ int d_disc(float t, float lo, float hi) {
  float u = (t - lo) / (hi - lo) * 128.f - 0.5f;
  int i = (int)rintf(u);
  return i < 0 ? 0 : (i > 127 ? 127 : i);
}

__global__ void k_face_feat(const float* __restrict__ verts, const int* __restrict__ faces,
                            const float* __restrict__ ce, const float* __restrict__ ae,
                            const float* __restrict__ re, const float* __restrict__ ne,
                            f16* __restrict__ emb)
{
  int face = blockIdx.x * 8 + (threadIdx.x >> 5);
  int lane = threadIdx.x & 31;
  if (face >= cM) return;
  int b = face / cNF;
  const int* f = faces + (long long)face * 3;
  float fc[3][3];
#pragma unroll
  for (int j = 0; j < 3; ++j) {
    int vi = f[j];
    const float* vp = verts + ((long long)b * cNV + vi) * 3;
    fc[j][0] = vp[0]; fc[j][1] = vp[1]; fc[j][2] = vp[2];
  }
  const int shj[3] = {2, 0, 1};
  float fcn[3][3];
#pragma unroll
  for (int j = 0; j < 3; ++j) {
    float nrm = fmaxf(sqrtf(fc[j][0]*fc[j][0] + fc[j][1]*fc[j][1] + fc[j][2]*fc[j][2]), 1e-12f);
    fcn[j][0] = fc[j][0] / nrm; fcn[j][1] = fc[j][1] / nrm; fcn[j][2] = fc[j][2] / nrm;
  }
  int idx[16];
#pragma unroll
  for (int j = 0; j < 3; ++j)
#pragma unroll
    for (int a = 0; a < 3; ++a) idx[j * 3 + a] = d_disc(fc[j][a], -1.f, 1.f);
#pragma unroll
  for (int j = 0; j < 3; ++j) {
    int s = shj[j];
    float c = fcn[j][0]*fcn[s][0] + fcn[j][1]*fcn[s][1] + fcn[j][2]*fcn[s][2];
    c = fminf(fmaxf(c, -1.f + 1e-5f), 1.f - 1e-5f);
    idx[9 + j] = d_disc(acosf(c), 0.f, cPI);
  }
  float e0[3], e1[3];
#pragma unroll
  for (int a = 0; a < 3; ++a) { e0[a] = fc[0][a] - fc[2][a]; e1[a] = fc[1][a] - fc[0][a]; }
  float cr[3] = { e0[1]*e1[2] - e0[2]*e1[1], e0[2]*e1[0] - e0[0]*e1[2], e0[0]*e1[1] - e0[1]*e1[0] };
  float cl = fmaxf(sqrtf(cr[0]*cr[0] + cr[1]*cr[1] + cr[2]*cr[2]), 1e-12f);
  float nv[3] = { cr[0]/cl, cr[1]/cl, cr[2]/cl };
  float area = sqrtf(nv[0]*nv[0] + nv[1]*nv[1] + nv[2]*nv[2]) * 0.5f;
  idx[12] = d_disc(area, 0.f, 4.f);
#pragma unroll
  for (int a = 0; a < 3; ++a) idx[13 + a] = d_disc(nv[a], -1.f, 1.f);

  long long base = (long long)face * 832;
  for (int p = 0; p < 9; ++p)
    for (int c = lane; c < 64; c += 32) emb[base + p*64 + c] = (f16)ce[idx[p]*64 + c];
  for (int p = 0; p < 3; ++p)
    for (int c = lane; c < 16; c += 32) emb[base + 576 + p*16 + c] = (f16)ae[idx[9+p]*16 + c];
  for (int c = lane; c < 16; c += 32)   emb[base + 624 + c] = (f16)re[idx[12]*16 + c];
  for (int p = 0; p < 3; ++p)
    for (int c = lane; c < 64; c += 32) emb[base + 640 + p*64 + c] = (f16)ne[idx[13+p]*64 + c];
}

// ---------------------------------------------------------------------------
// graph kernels
// ---------------------------------------------------------------------------
__global__ void k_deg(const int* __restrict__ fe, float* __restrict__ deg) {
  int i = blockIdx.x * blockDim.x + threadIdx.x;
  if (i >= cB * cNE) return;
  int b = i / cNE;
  atomicAdd(&deg[fe[(long long)i * 2 + 1] + b * cNF], 1.f);
}

__global__ void k_edge_gather(const int* __restrict__ fe, const float* __restrict__ xp,
                              float* __restrict__ agg, int C) {
  long long total = (long long)cB * cNE * C;
  long long i = (long long)blockIdx.x * blockDim.x + threadIdx.x;
  long long s = (long long)gridDim.x * blockDim.x;
  for (; i < total; i += s) {
    long long e = i / C; int c = (int)(i % C);
    int b = (int)(e / cNE);
    int src = fe[e*2]   + b * cNF;
    int dst = fe[e*2+1] + b * cNF;
    atomicAdd(&agg[(long long)dst * C + c], xp[(long long)src * C + c]);
  }
}

__global__ void k_div_deg(const float* __restrict__ agg, const float* __restrict__ deg,
                          f16* __restrict__ out, int C) {
  long long total = (long long)cM * C;
  long long i = (long long)blockIdx.x * blockDim.x + threadIdx.x;
  long long s = (long long)gridDim.x * blockDim.x;
  for (; i < total; i += s) {
    int row = (int)(i / C);
    out[i] = (f16)(agg[i] / fmaxf(deg[row], 1.f));
  }
}

// ---------------------------------------------------------------------------
// row-wise L2 norm and/or silu+LayerNorm (wave32 per row)
// ---------------------------------------------------------------------------
__global__ void k_rownorm(float* __restrict__ x, f16* __restrict__ x16,
                          const float* __restrict__ g, const float* __restrict__ bb,
                          int M, int C, int mode)
{
  int row = blockIdx.x * 8 + (threadIdx.x >> 5);
  int lane = threadIdx.x & 31;
  if (row >= M) return;
  float v[18];
  const int nt = (C + 31) / 32;
  float ss = 0.f;
  for (int t = 0; t < nt; ++t) {
    int c = t * 32 + lane;
    float u = (c < C) ? x[(long long)row * C + c] : 0.f;
    v[t] = u; ss += u * u;
  }
  if (mode & RN_L2) {
    for (int o = 16; o > 0; o >>= 1) ss += __shfl_xor(ss, o, 32);
    float sc = 1.f / fmaxf(sqrtf(ss), 1e-12f);
    for (int t = 0; t < nt; ++t) v[t] *= sc;
  }
  if (mode & RN_SILULN) {
    float s = 0.f;
    for (int t = 0; t < nt; ++t) {
      int c = t * 32 + lane;
      float u = (c < C) ? v[t] / (1.f + __expf(-v[t])) : 0.f;
      v[t] = u; s += u;
    }
    for (int o = 16; o > 0; o >>= 1) s += __shfl_xor(s, o, 32);
    float mu = s / C;
    float vs = 0.f;
    for (int t = 0; t < nt; ++t) {
      int c = t * 32 + lane;
      float d = (c < C) ? v[t] - mu : 0.f;
      vs += d * d;
    }
    for (int o = 16; o > 0; o >>= 1) vs += __shfl_xor(vs, o, 32);
    float inv = rsqrtf(vs / C + 1e-5f);
    for (int t = 0; t < nt; ++t) {
      int c = t * 32 + lane;
      if (c < C) v[t] = (v[t] - mu) * inv * g[c] + bb[c];
    }
  }
  for (int t = 0; t < nt; ++t) {
    int c = t * 32 + lane;
    if (c < C) {
      long long i = (long long)row * C + c;
      x[i] = v[t];
      if (x16) x16[i] = (f16)v[t];
    }
  }
}

// ---------------------------------------------------------------------------
// vertex scatter-mean + LFQ helpers
// ---------------------------------------------------------------------------
__global__ void k_scatter_codes(const float* __restrict__ Y, const int* __restrict__ faces,
                                float* __restrict__ vsum, float* __restrict__ vcnt) {
  long long total = (long long)cM * 3 * 192;
  long long i = (long long)blockIdx.x * blockDim.x + threadIdx.x;
  long long s = (long long)gridDim.x * blockDim.x;
  for (; i < total; i += s) {
    int c = (int)(i % 192);
    long long fj = i / 192;
    int j = (int)(fj % 3);
    int fface = (int)(fj / 3);
    int b = fface / cNF;
    int vi = faces[(long long)fface * 3 + j] + b * cNV;
    atomicAdd(&vsum[(long long)vi * 192 + c], Y[(long long)fface * 576 + j * 192 + c]);
    if (c == 0) atomicAdd(&vcnt[vi], 1.f);
  }
}

__global__ void k_vmean(const float* __restrict__ vsum, const float* __restrict__ vcnt,
                        float* __restrict__ resid, f16* __restrict__ resid16) {
  long long total = (long long)cMV * 192;
  long long i = (long long)blockIdx.x * blockDim.x + threadIdx.x;
  long long s = (long long)gridDim.x * blockDim.x;
  for (; i < total; i += s) {
    int row = (int)(i / 192);
    float v = vsum[i] / fmaxf(vcnt[row], 1e-5f);
    resid[i] = v; resid16[i] = (f16)v;
  }
}

// sign codes, zero-padded K: 14 -> 32 so the out-projection GEMM is K-aligned
__global__ void k_sign(const float* __restrict__ h, f16* __restrict__ qh) {
  long long total = (long long)cMV * 32;
  long long i = (long long)blockIdx.x * blockDim.x + threadIdx.x;
  long long s = (long long)gridDim.x * blockDim.x;
  for (; i < total; i += s) {
    int c = (int)(i % 32);
    long long r = i / 32;
    qh[i] = (c < 14) ? (f16)(h[r * 14 + c] > 0.f ? 1.f : -1.f) : (f16)0.f;
  }
}

__global__ void k_lfq_update(float* __restrict__ resid, f16* __restrict__ resid16,
                             float* __restrict__ quant, const float* __restrict__ q, long long n) {
  long long i = (long long)blockIdx.x * blockDim.x + threadIdx.x;
  long long s = (long long)gridDim.x * blockDim.x;
  for (; i < n; i += s) {
    float r = resid[i] - q[i];
    resid[i] = r; resid16[i] = (f16)r;
    quant[i] += q[i];
  }
}

__global__ void k_gather_fq(const float* __restrict__ quant, const int* __restrict__ faces,
                            f16* __restrict__ fq) {
  long long total = (long long)cM * 576;
  long long i = (long long)blockIdx.x * blockDim.x + threadIdx.x;
  long long s = (long long)gridDim.x * blockDim.x;
  for (; i < total; i += s) {
    int r = (int)(i % 576);
    int fface = (int)(i / 576);
    int j = r / 192, c = r % 192;
    int b = fface / cNF;
    int vi = faces[(long long)fface * 3 + j] + b * cNV;
    fq[i] = (f16)quant[(long long)vi * 192 + c];
  }
}

// ---------------------------------------------------------------------------
// decoder group-norm (stats over channel-group x full sequence), SE, combine
// ---------------------------------------------------------------------------
__global__ void k_gn_stats(const float* __restrict__ u, float* __restrict__ stats, int C, int G) {
  int b = blockIdx.x / G, g = blockIdx.x % G;
  int cg = C / G;
  long long cnt = (long long)cg * cNF;
  float s = 0.f, ss = 0.f;
  for (long long i = threadIdx.x; i < cnt; i += blockDim.x) {
    int cc = (int)(i / cNF);
    long long n = i % cNF;
    float x = u[((long long)b * cNF + n) * C + g * cg + cc];
    s += x; ss += x * x;
  }
  __shared__ float sh0[256], sh1[256];
  sh0[threadIdx.x] = s; sh1[threadIdx.x] = ss; __syncthreads();
  for (int o = 128; o > 0; o >>= 1) {
    if ((int)threadIdx.x < o) { sh0[threadIdx.x] += sh0[threadIdx.x + o]; sh1[threadIdx.x] += sh1[threadIdx.x + o]; }
    __syncthreads();
  }
  if (threadIdx.x == 0) {
    float m = sh0[0] / (float)cnt;
    stats[blockIdx.x * 2]     = m;
    stats[blockIdx.x * 2 + 1] = sh1[0] / (float)cnt - m * m;
  }
}

__global__ void k_gn_apply(float* __restrict__ u, f16* __restrict__ u16,
                           const float* __restrict__ stats,
                           const float* __restrict__ g, const float* __restrict__ bb,
                           int C, int G) {
  long long total = (long long)cM * C;
  long long i = (long long)blockIdx.x * blockDim.x + threadIdx.x;
  long long s = (long long)gridDim.x * blockDim.x;
  int cg = C / G;
  for (; i < total; i += s) {
    int c = (int)(i % C);
    int row = (int)(i / C);
    int b = row / cNF;
    int gi = b * G + c / cg;
    float mu = stats[gi * 2], var = stats[gi * 2 + 1];
    float y = (u[i] - mu) * rsqrtf(var + 1e-5f) * g[c] + bb[c];
    y = y / (1.f + __expf(-y));                 // fused silu
    u[i] = y;
    if (u16) u16[i] = (f16)y;
  }
}

__global__ void k_se_mean(const float* __restrict__ u, float* __restrict__ m, int C) {
  int b = blockIdx.x / C, c = blockIdx.x % C;
  float s = 0.f;
  for (int n = threadIdx.x; n < cNF; n += blockDim.x)
    s += u[((long long)b * cNF + n) * C + c];
  __shared__ float sh[256];
  sh[threadIdx.x] = s; __syncthreads();
  for (int o = 128; o > 0; o >>= 1) {
    if ((int)threadIdx.x < o) sh[threadIdx.x] += sh[threadIdx.x + o];
    __syncthreads();
  }
  if (threadIdx.x == 0) m[blockIdx.x] = sh[0] / (float)cNF;
}

__global__ void k_se1(const float* __restrict__ m, const float* __restrict__ w1,
                      const float* __restrict__ b1, float* __restrict__ s, int C, int D) {
  int i = blockIdx.x * blockDim.x + threadIdx.x;
  if (i >= cB * D) return;
  int b = i / D, d = i % D;
  float a = b1[d];
  for (int c = 0; c < C; ++c) a += m[b * C + c] * w1[c * D + d];
  s[i] = a / (1.f + __expf(-a));
}

__global__ void k_se2(const float* __restrict__ s, const float* __restrict__ w2,
                      const float* __restrict__ b2, float* __restrict__ gate, int C, int D) {
  int i = blockIdx.x * blockDim.x + threadIdx.x;
  if (i >= cB * C) return;
  int b = i / C, c = i % C;
  float a = b2[c];
  for (int d = 0; d < D; ++d) a += s[b * D + d] * w2[d * C + c];
  gate[i] = 1.f / (1.f + __expf(-a));
}

__global__ void k_combine(const float* __restrict__ u, const float* __restrict__ gate,
                          const float* __restrict__ R, float* __restrict__ h,
                          f16* __restrict__ h16, int C) {
  long long total = (long long)cM * C;
  long long i = (long long)blockIdx.x * blockDim.x + threadIdx.x;
  long long s = (long long)gridDim.x * blockDim.x;
  for (; i < total; i += s) {
    int c = (int)(i % C);
    int row = (int)(i / C);
    int b = row / cNF;
    float y = u[i] * gate[b * C + c] + R[i];
    h[i] = y; h16[i] = (f16)y;
  }
}

// ---------------------------------------------------------------------------
// host orchestration
// ---------------------------------------------------------------------------
extern "C" void kernel_launch(void* const* d_in, const int* in_sizes, int n_in,
                              void* d_out, int out_size, void* d_ws, size_t ws_size,
                              hipStream_t stream) {
  (void)in_sizes; (void)n_in; (void)out_size; (void)ws_size;
  auto F  = [&](int i) { return (const float*)d_in[i]; };
  auto Ii = [&](int i) { return (const int*)d_in[i]; };

  const float* vertices = F(0);
  const int*   faces    = Ii(1);
  const int*   fedges   = Ii(2);

  // --- workspace bump allocator ---
  size_t off = 0;
  auto alloc = [&](size_t bytes) -> void* {
    off = (off + 255) & ~(size_t)255;
    void* p = (char*)d_ws + off;
    off += bytes;
    return p;
  };

  const int dins[5]  = {192, 64, 128, 256, 256};
  const int douts[5] = {64, 128, 256, 256, 576};

  // f16 transposed weights [N][K]
  f16* w_pin = (f16*)alloc((size_t)192 * 832 * 2);
  f16 *w_sp[5], *w_sl[5], *w_sr[5];
  for (int i = 0; i < 5; ++i) {
    w_sp[i] = (f16*)alloc((size_t)dins[i] * dins[i] * 2);
    w_sl[i] = (f16*)alloc((size_t)douts[i] * dins[i] * 2);
    w_sr[i] = (f16*)alloc((size_t)douts[i] * dins[i] * 2);
  }
  f16* w_cb = (f16*)alloc((size_t)576 * 576 * 2);
  f16 *w_qi[2], *w_qo[2];
  for (int q = 0; q < 2; ++q) {
    w_qi[q] = (f16*)alloc((size_t)16 * 192 * 2);    // [Npad=16][K=192]
    w_qo[q] = (f16*)alloc((size_t)192 * 32 * 2);    // [N=192][Kpad=32]
  }
  f16* w_dec = (f16*)alloc((size_t)7 * 128 * 576 * 2);   // [T][O][I]
  const int bdin[3] = {128, 128, 192}, bdout[3] = {128, 192, 256};
  const int bdse[3] = {32, 48, 64};
  const int hasres[3] = {0, 1, 1};
  const int bbase[3] = {50, 62, 76};
  f16 *w_c1[3], *w_c2[3], *w_rw[3] = {nullptr, nullptr, nullptr};
  for (int k = 0; k < 3; ++k) {
    w_c1[k] = (f16*)alloc((size_t)3 * bdout[k] * bdin[k] * 2);
    w_c2[k] = (f16*)alloc((size_t)3 * bdout[k] * bdout[k] * 2);
    if (hasres[k]) w_rw[k] = (f16*)alloc((size_t)bdout[k] * bdin[k] * 2);
  }
  f16* w_log = (f16*)alloc((size_t)1152 * 256 * 2);

  // activation scratch
  f16*   emb16 = (f16*)alloc((size_t)cM * 832 * 2);   // later reused as fq16
  float* F0 = (float*)alloc((size_t)cM * 576 * 4);
  float* F1 = (float*)alloc((size_t)cM * 576 * 4);
  float* F2 = (float*)alloc((size_t)cM * 576 * 4);
  f16*  X16a = (f16*)alloc((size_t)cM * 576 * 2);
  f16*  X16b = (f16*)alloc((size_t)cM * 576 * 2);
  float* deg   = (float*)alloc((size_t)cM * 4);
  float* vsum  = (float*)alloc((size_t)cMV * 192 * 4);
  float* vcnt  = (float*)alloc((size_t)cMV * 4);
  float* resid = (float*)alloc((size_t)cMV * 192 * 4);
  f16*   resid16 = (f16*)alloc((size_t)cMV * 192 * 2);
  float* quant = (float*)alloc((size_t)cMV * 192 * 4);
  float* hq    = (float*)alloc((size_t)cMV * 14 * 4);
  f16*   qh16  = (f16*)alloc((size_t)cMV * 32 * 2);   // K-padded sign codes
  float* qbuf  = (float*)alloc((size_t)cMV * 192 * 4);
  float* stats = (float*)alloc((size_t)16 * 2 * 4);
  float* semean = (float*)alloc((size_t)cB * 256 * 4);
  float* sehid  = (float*)alloc((size_t)cB * 64 * 4);
  float* gate   = (float*)alloc((size_t)cB * 256 * 4);

  auto zero = [&](float* p, long long n) {
    int g = (int)((n + 255) / 256); if (g > 8192) g = 8192;
    k_zero<<<g, 256, 0, stream>>>(p, n);
  };
  auto wt = [&](const float* s, f16* d, int K, int N, int Kpad, int Npad) {
    long long n = (long long)Kpad * Npad;
    int g = (int)((n + 255) / 256); if (g > 4096) g = 4096;
    k_wt<<<g, 256, 0, stream>>>(s, d, K, N, Kpad, Npad);
  };
  auto convwt = [&](const float* s, f16* d, int O, int I, int T) {
    long long n = (long long)O * I * T;
    int g = (int)((n + 255) / 256); if (g > 4096) g = 4096;
    k_convwt<<<g, 256, 0, stream>>>(s, d, O, I, T);
  };
  auto gemm = [&](const f16* A, const f16* Bt, const float* bias, float* C, f16* C16,
                  int M, int N, int K, int shift, int seqLen, int flags) {
    dim3 g((N + 63) / 64, (M + 127) / 128);
    k_gemm<<<g, 128, 0, stream>>>(A, Bt, bias, C, C16, M, N, K, shift, seqLen, flags);
  };

  // --- weight repack to transposed f16 ---
  wt(F(7), w_pin, 832, 192, 832, 192);
  for (int i = 0; i < 5; ++i) {
    wt(F(9 + 5*i),  w_sp[i], dins[i], dins[i],  dins[i], dins[i]);
    wt(F(11 + 5*i), w_sl[i], dins[i], douts[i], dins[i], douts[i]);
    wt(F(13 + 5*i), w_sr[i], dins[i], douts[i], dins[i], douts[i]);
  }
  wt(F(36), w_cb, 576, 576, 576, 576);
  for (int q = 0; q < 2; ++q) {
    wt(F(38 + 4*q),     w_qi[q], 192, 14, 192, 16);   // pad N 14->16
    wt(F(38 + 4*q + 2), w_qo[q], 14, 192, 32, 192);   // pad K 14->32
  }
  convwt(F(46), w_dec, 128, 576, 7);
  for (int k = 0; k < 3; ++k) {
    convwt(F(bbase[k] + 0), w_c1[k], bdout[k], bdin[k], 3);
    convwt(F(bbase[k] + 4), w_c2[k], bdout[k], bdout[k], 3);
    if (hasres[k]) convwt(F(bbase[k] + 12), w_rw[k], bdout[k], bdin[k], 1);
  }
  wt(F(90), w_log, 256, 1152, 256, 1152);

  // --- 1. face features -> emb16 [cM,832] ---
  k_face_feat<<<cM / 8, 256, 0, stream>>>(vertices, faces, F(3), F(4), F(5), F(6), emb16);

  // --- 2. proj_in: x = emb @ W + b  -> f16 X16a [cM,192] ---
  gemm(emb16, w_pin, F(8), F1, X16a, cM, 192, 832, 0, 0, 0);

  // --- 3. SAGE encoder ---
  zero(deg, cM);
  k_deg<<<(cB * cNE + 255) / 256, 256, 0, stream>>>(fedges, deg);
  for (int i = 0; i < 5; ++i) {
    const int di = dins[i], dn = douts[i];
    // xp = relu(x @ proj_w + proj_b)
    gemm(X16a, w_sp[i], F(10 + 5*i), F1, nullptr, cM, di, di, 0, 0, GF_RELU);
    // agg = segment_sum(xp[src]) / deg
    zero(F2, (long long)cM * di);
    k_edge_gather<<<8192, 256, 0, stream>>>(fedges, F1, F2, di);
    k_div_deg<<<8192, 256, 0, stream>>>(F2, deg, X16b, di);
    // x = l2norm(agg @ l_w + l_b + x @ r_w)   (+ silu/LN on layer 0)
    gemm(X16b, w_sl[i], F(12 + 5*i), F1, nullptr, cM, dn, di, 0, 0, 0);
    gemm(X16a, w_sr[i], nullptr,     F1, nullptr, cM, dn, di, 0, 0, GF_ACC);
    k_rownorm<<<cM / 8, 256, 0, stream>>>(F1, X16a, F(34), F(35), cM, dn,
                                          (i == 0) ? (RN_L2 | RN_SILULN) : RN_L2);
  }

  // --- 4. proj_cb + vertex scatter-mean ---
  gemm(X16a, w_cb, F(37), F2, nullptr, cM, 576, 576, 0, 0, 0);
  zero(vsum, (long long)cMV * 192);
  zero(vcnt, cMV);
  k_scatter_codes<<<8192, 256, 0, stream>>>(F2, faces, vsum, vcnt);
  k_vmean<<<4096, 256, 0, stream>>>(vsum, vcnt, resid, resid16);
  zero(quant, (long long)cMV * 192);

  // --- 5. residual LFQ (2 quantizers) ---
  for (int q = 0; q < 2; ++q) {
    gemm(resid16, w_qi[q], F(39 + 4*q), hq, nullptr, cMV, 14, 192, 0, 0, 0);
    k_sign<<<2048, 256, 0, stream>>>(hq, qh16);
    gemm(qh16, w_qo[q], F(41 + 4*q), qbuf, nullptr, cMV, 192, 32, 0, 0, 0);
    k_lfq_update<<<4096, 256, 0, stream>>>(resid, resid16, quant, qbuf, (long long)cMV * 192);
  }

  // --- 6. gather quantized codes per face -> fq16 [cM,576] (reuse emb16) ---
  f16* fq16 = emb16;
  k_gather_fq<<<8192, 256, 0, stream>>>(quant, faces, fq16);

  // --- 7. decoder init conv (k=7, pad=3) as 7 shifted GEMMs, then silu + LN ---
  for (int t = 0; t < 7; ++t)
    gemm(fq16, w_dec + (size_t)t * 128 * 576, (t == 0) ? F(47) : nullptr,
         F0, nullptr, cM, 128, 576, t - 3, cNF, (t == 0) ? 0 : GF_ACC);
  k_rownorm<<<cM / 8, 256, 0, stream>>>(F0, X16a, F(48), F(49), cM, 128, RN_SILULN);

  // --- 8. residual conv blocks ---
  for (int k = 0; k < 3; ++k) {
    const int di = bdin[k], dn = bdout[k], ds = bdse[k], pb = bbase[k];
    float* R = F0;
    if (hasres[k]) {
      gemm(X16a, w_rw[k], F(pb + 13), F2, nullptr, cM, dn, di, 0, 0, 0);
      R = F2;
    }
    // u1 = silu(gn1(conv3(h)))
    for (int t = 0; t < 3; ++t)
      gemm(X16a, w_c1[k] + (size_t)t * dn * di, (t == 0) ? F(pb + 1) : nullptr,
           F1, nullptr, cM, dn, di, t - 1, cNF, (t == 0) ? 0 : GF_ACC);
    k_gn_stats<<<cB * 8, 256, 0, stream>>>(F1, stats, dn, 8);
    k_gn_apply<<<8192, 256, 0, stream>>>(F1, X16b, stats, F(pb + 2), F(pb + 3), dn, 8);
    // u2 = silu(gn2(conv3(u1)))
    for (int t = 0; t < 3; ++t)
      gemm(X16b, w_c2[k] + (size_t)t * dn * dn, (t == 0) ? F(pb + 5) : nullptr,
           F1, nullptr, cM, dn, dn, t - 1, cNF, (t == 0) ? 0 : GF_ACC);
    k_gn_stats<<<cB * 8, 256, 0, stream>>>(F1, stats, dn, 8);
    k_gn_apply<<<8192, 256, 0, stream>>>(F1, nullptr, stats, F(pb + 6), F(pb + 7), dn, 8);
    // SE gate
    k_se_mean<<<cB * dn, 256, 0, stream>>>(F1, semean, dn);
    k_se1<<<(cB * ds + 255) / 256, 256, 0, stream>>>(semean, F(pb + 8), F(pb + 9), sehid, dn, ds);
    k_se2<<<(cB * dn + 255) / 256, 256, 0, stream>>>(sehid, F(pb + 10), F(pb + 11), gate, dn, ds);
    // h = u2 * gate + res
    k_combine<<<8192, 256, 0, stream>>>(F1, gate, R, F0, X16a, dn);
  }

  // --- 9. logits: out[cM,1152] = h @ W + b ---
  gemm(X16a, w_log, F(91), (float*)d_out, nullptr, cM, 1152, 256, 0, 0, 0);
}
